// VariationalRNNEncoder_65481071394913
// MI455X (gfx1250) — matrix operand order repaired and checked by reading, not compiled, not run on hardware
//
#include <hip/hip_runtime.h>
#include <cstdint>
#include <cstddef>

#define DEV __device__ __forceinline__

typedef __bf16 bf16;
typedef __attribute__((ext_vector_type(16))) __bf16 v16bf;
typedef __attribute__((ext_vector_type(8)))  float  v8f;

namespace {

constexpr int B_ = 256, T_ = 512, D_ = 64, H_ = 512, Z_ = 64;
constexpr int TB = T_ * B_;

struct Frag32B { uint4 lo, hi; };

DEV bf16 f2bf(float f) {
  unsigned u = __builtin_bit_cast(unsigned, f);
  unsigned r = (u + 0x7FFFu + ((u >> 16) & 1u)) >> 16;
  unsigned short s = (unsigned short)r;
  return __builtin_bit_cast(bf16, s);
}

DEV float softplus_f(float x) { return x > 20.f ? x : log1pf(expf(x)); }

// A fragment: 16x32 (MxK) bf16. Lane L (r = L&15, kh = L>>4) holds row m0+r,
// K = k0+16kh .. +15 (contiguous, matches B fragment)
DEV v16bf load_a_frag(const bf16* __restrict__ A, int lda, int m0, int k0, int lane) {
  int r = lane & 15, kh = lane >> 4;
  const bf16* p = A + (size_t)(m0 + r) * lda + k0 + 16 * kh;
  Frag32B f;
  f.lo = *(const uint4*)p;
  f.hi = *(const uint4*)(p + 8);
  return __builtin_bit_cast(v16bf, f);
}

// B fragment: 32x16 (KxN) bf16, from row-major W[N,K] (C = A*W^T). Lane L
// (n = L&15, half = L>>4) holds col n0+n, K = k0+16*half .. +15 (contiguous).
DEV v16bf load_b_frag(const bf16* __restrict__ W, int ldw, int n0, int k0, int lane) {
  int n = lane & 15, half = lane >> 4;
  const bf16* p = W + (size_t)(n0 + n) * ldw + k0 + 16 * half;
  Frag32B f;
  f.lo = *(const uint4*)p;
  f.hi = *(const uint4*)(p + 8);
  return __builtin_bit_cast(v16bf, f);
}

// Generic GEMM: C[M,N] = act(concatK(A0,A1) @ W^T + bias).
// Block = 128 thr (4 waves), tile 128M x 64N; wave = 32 rows x 64 cols
// (2 A-frags, 8 accumulators). B (weights) staged in LDS in fragment order,
// double-buffered, one barrier per 32-K step.
// EPI: 0 = linear (f32 out), 1 = relu (bf16 out + optional f32 out).
template <int EPI>
__global__ __launch_bounds__(128) void gemm_bf16_kernel(
    const bf16* __restrict__ A0, const bf16* __restrict__ A1, int K0, int K,
    int lda0, int lda1, const bf16* __restrict__ W, int ldw,
    const float* __restrict__ bias, float* __restrict__ Cf,
    bf16* __restrict__ Cb, int ldc) {
  __shared__ __align__(16) bf16 shB[2][64 * 32]; // 2 x 4KB, fragment-ordered

  const int tid = threadIdx.x;
  const int lane = tid & 31;
  const int wave = tid >> 5;
  const int m0 = blockIdx.x * 128 + wave * 32;
  const int n0 = blockIdx.y * 64;

  // This thread stages one 32B B fragment: subtile sj = tid>>5, its own lane.
  const int sj = tid >> 5;
  const int sn = n0 + 16 * sj + (lane & 15);
  const int skh = lane >> 4;

  auto stage = [&](int buf, int k0) {
    const bf16* p = W + (size_t)sn * ldw + k0 + 16 * skh;
    uint4 lo = *(const uint4*)p;
    uint4 hi = *(const uint4*)(p + 8);
    uint4* d = (uint4*)&shB[buf][tid * 16];
    d[0] = lo;
    d[1] = hi;
  };

  v8f acc[2][4] = {};
  stage(0, 0);
  __syncthreads();
  int buf = 0;
  for (int k0 = 0; k0 < K; k0 += 32) {
    const bf16* A; int kk, lda;
    if (k0 < K0) { A = A0; kk = k0;      lda = lda0; }
    else         { A = A1; kk = k0 - K0; lda = lda1; }
    if (k0 + 32 < K) stage(buf ^ 1, k0 + 32);
    v16bf a0 = load_a_frag(A, lda, m0,      kk, lane);
    v16bf a1 = load_a_frag(A, lda, m0 + 16, kk, lane);
#pragma unroll
    for (int j = 0; j < 4; ++j) {
      const uint4* s = (const uint4*)&shB[buf][(j * 32 + lane) * 16];
      Frag32B fb;
      fb.lo = s[0];
      fb.hi = s[1];
      v16bf b = __builtin_bit_cast(v16bf, fb);
      acc[0][j] = __builtin_amdgcn_wmma_f32_16x16x32_bf16(
          false, a0, false, b, (short)0, acc[0][j], false, false);
      acc[1][j] = __builtin_amdgcn_wmma_f32_16x16x32_bf16(
          false, a1, false, b, (short)0, acc[1][j], false, false);
    }
    __syncthreads();
    buf ^= 1;
  }

  const int ccol = lane & 15;
  const int rhalf = lane >> 4;
#pragma unroll
  for (int mm = 0; mm < 2; ++mm) {
#pragma unroll
    for (int j = 0; j < 4; ++j) {
#pragma unroll
      for (int i = 0; i < 8; ++i) {
        int row = m0 + 16 * mm + i + 8 * rhalf;
        int col = n0 + 16 * j + ccol;
        float v = acc[mm][j][i];
        if (bias) v += bias[col];
        if (EPI == 1) v = v > 0.f ? v : 0.f;
        size_t idx = (size_t)row * ldc + col;
        if (Cf) Cf[idx] = v;
        if (Cb) Cb[idx] = f2bf(v);
      }
    }
  }
}

// Head GEMM: A[256,512] @ Wpacked[128,512]^T; cols 0..63 -> mean (f32),
// cols 64..127 -> softplus -> std (f32). grid (4, 2), wave = 16 rows.
__global__ __launch_bounds__(128) void gemm_heads_kernel(
    const bf16* __restrict__ A, const bf16* __restrict__ W,
    const float* __restrict__ bMean, const float* __restrict__ bStd,
    float* __restrict__ outMean, float* __restrict__ outStd) {
  const int lane = threadIdx.x & 31;
  const int wave = threadIdx.x >> 5;
  const int m0 = blockIdx.x * 64 + wave * 16;
  const int n0 = blockIdx.y * 64;

  v8f acc[4] = {};
  for (int k0 = 0; k0 < 512; k0 += 32) {
    v16bf a = load_a_frag(A, 512, m0, k0, lane);
#pragma unroll
    for (int j = 0; j < 4; ++j) {
      v16bf b = load_b_frag(W, 512, n0 + 16 * j, k0, lane);
      acc[j] = __builtin_amdgcn_wmma_f32_16x16x32_bf16(
          false, a, false, b, (short)0, acc[j], false, false);
    }
  }
  const int ccol = lane & 15;
  const int rhalf = lane >> 4;
#pragma unroll
  for (int j = 0; j < 4; ++j) {
#pragma unroll
    for (int i = 0; i < 8; ++i) {
      int row = m0 + i + 8 * rhalf;
      int col = n0 + 16 * j + ccol;
      float v = acc[j][i];
      if (col < 64) {
        outMean[row * 64 + col] = v + bMean[col];
      } else {
        int c = col - 64;
        outStd[row * 64 + c] = softplus_f(v + bStd[c]);
      }
    }
  }
}

__global__ void conv_bf16_kernel(const float* __restrict__ s, bf16* __restrict__ d, int n) {
  int i = blockIdx.x * blockDim.x + threadIdx.x;
  if (i < n) d[i] = f2bf(s[i]);
}

// x[B,T,D] -> xb[T,B,D] bf16
__global__ void xconv_kernel(const float* __restrict__ x, bf16* __restrict__ xb) {
  int i = blockIdx.x * blockDim.x + threadIdx.x; // [t][b][d]
  int d = i & 63;
  int b = (i >> 6) & 255;
  int t = i >> 14;
  xb[i] = f2bf(x[(size_t)b * (T_ * D_) + t * D_ + d]);
}

__global__ void zero_h_kernel(float* __restrict__ h, bf16* __restrict__ hb) {
  int i = blockIdx.x * blockDim.x + threadIdx.x; // B*H
  h[i] = 0.f;
  hb[i] = f2bf(0.f);
}

// z = eps * std + mean -> bf16
__global__ void zcalc_kernel(const float* __restrict__ em, const float* __restrict__ es,
                             const float* __restrict__ eps, bf16* __restrict__ zb) {
  int i = blockIdx.x * blockDim.x + threadIdx.x; // B*Z
  zb[i] = f2bf(eps[i] * es[i] + em[i]);
}

// GRU cell: gates in torch order r,z,n over 1536 cols of Gi/Gh
__global__ void gru_cell_kernel(const float* __restrict__ Gi, const float* __restrict__ Gh,
                                float* __restrict__ h, bf16* __restrict__ hb) {
  int i = blockIdx.x * blockDim.x + threadIdx.x; // B*H
  int b = i >> 9;
  int n = i & 511;
  const float* gi = Gi + (size_t)b * 1536;
  const float* gh = Gh + (size_t)b * 1536;
  float r  = 1.f / (1.f + expf(-(gi[n]        + gh[n])));
  float zg = 1.f / (1.f + expf(-(gi[n + 512]  + gh[n + 512])));
  float nn = tanhf(gi[n + 1024] + r * gh[n + 1024]);
  float hv = h[i];
  float hn = (1.f - zg) * nn + zg * hv;
  h[i] = hn;
  hb[i] = f2bf(hn);
}

// res = [phi_z_fin (f32), h_fin]
__global__ void final_kernel(const float* __restrict__ pzf, const float* __restrict__ h,
                             float* __restrict__ res) {
  int i = blockIdx.x * blockDim.x + threadIdx.x; // B*2H
  int b = i >> 10;
  int j = i & 1023;
  res[i] = (j < 512) ? pzf[(size_t)b * 512 + j] : h[(size_t)b * 512 + (j - 512)];
}

} // namespace

extern "C" void kernel_launch(void* const* d_in, const int* in_sizes, int n_in,
                              void* d_out, int out_size, void* d_ws, size_t ws_size,
                              hipStream_t stream) {
  (void)in_sizes; (void)n_in; (void)out_size; (void)ws_size;

  const float* x        = (const float*)d_in[0];   // [B,T,D]
  const float* eps      = (const float*)d_in[1];   // [T,B,Z]
  const float* phi_x_W1 = (const float*)d_in[2];
  const float* phi_x_b1 = (const float*)d_in[3];
  const float* phi_x_W2 = (const float*)d_in[4];
  const float* phi_x_b2 = (const float*)d_in[5];
  const float* phi_z_W  = (const float*)d_in[6];
  const float* phi_z_b  = (const float*)d_in[7];
  const float* enc_W1   = (const float*)d_in[8];
  const float* enc_b1   = (const float*)d_in[9];
  const float* enc_W2   = (const float*)d_in[10];
  const float* enc_b2   = (const float*)d_in[11];
  const float* enc_mean_W = (const float*)d_in[12];
  const float* enc_mean_b = (const float*)d_in[13];
  const float* enc_std_W  = (const float*)d_in[14];
  const float* enc_std_b  = (const float*)d_in[15];
  const float* prior_W    = (const float*)d_in[16];
  const float* prior_b    = (const float*)d_in[17];
  const float* prior_mean_W = (const float*)d_in[18];
  const float* prior_mean_b = (const float*)d_in[19];
  const float* prior_std_W  = (const float*)d_in[20];
  const float* prior_std_b  = (const float*)d_in[21];
  const float* gru_Wih    = (const float*)d_in[22]; // [1536,1024]
  const float* gru_Whh    = (const float*)d_in[23]; // [1536,512]

  // ---- workspace carve-up (256B aligned) ----
  char* w = (char*)d_ws;
  size_t off = 0;
  auto alloc = [&](size_t bytes) {
    void* p = w + off;
    off += (bytes + 255) & ~(size_t)255;
    return p;
  };
  bf16* PhiX = (bf16*)alloc((size_t)TB * H_ * 2);     // [T,B,H] precomputed phi_x
  bf16* H1   = (bf16*)alloc((size_t)TB * H_ * 2);     // phi_x hidden 1
  bf16* XB   = (bf16*)alloc((size_t)TB * D_ * 2);     // x transposed to [T,B,D]
  bf16* Wx1b = (bf16*)alloc(512 * 64 * 2);
  bf16* Wx2b = (bf16*)alloc(512 * 512 * 2);
  bf16* Wzb  = (bf16*)alloc(512 * 64 * 2);
  bf16* We1b = (bf16*)alloc(512 * 1024 * 2);
  bf16* We2b = (bf16*)alloc(512 * 512 * 2);
  bf16* Wheb = (bf16*)alloc(128 * 512 * 2);           // [enc_mean; enc_std]
  bf16* Wpb  = (bf16*)alloc(512 * 512 * 2);
  bf16* Whpb = (bf16*)alloc(128 * 512 * 2);           // [prior_mean; prior_std]
  bf16* Wihb = (bf16*)alloc(1536 * 1024 * 2);
  bf16* Whhb = (bf16*)alloc(1536 * 512 * 2);
  float* hF  = (float*)alloc((size_t)B_ * H_ * 4);
  bf16*  hB  = (bf16*)alloc((size_t)B_ * H_ * 2);
  bf16*  E1  = (bf16*)alloc((size_t)B_ * H_ * 2);
  bf16*  E2  = (bf16*)alloc((size_t)B_ * H_ * 2);
  bf16*  Pp  = (bf16*)alloc((size_t)B_ * H_ * 2);
  bf16*  PZ  = (bf16*)alloc((size_t)B_ * H_ * 2);
  float* PZf = (float*)alloc((size_t)B_ * H_ * 4);
  bf16*  Zb  = (bf16*)alloc((size_t)B_ * Z_ * 2);
  float* Gi  = (float*)alloc((size_t)B_ * 1536 * 4);
  float* Gh  = (float*)alloc((size_t)B_ * 1536 * 4);

  // ---- output carve-up: (res, em, es, pm, ps) ----
  float* out = (float*)d_out;
  const size_t TBZ = (size_t)T_ * B_ * Z_;
  float* res = out;
  float* em  = out + (size_t)B_ * 2 * H_;
  float* es  = em + TBZ;
  float* pm  = es + TBZ;
  float* ps  = pm + TBZ;

  auto conv = [&](const float* s, bf16* d, int n) {
    conv_bf16_kernel<<<(n + 255) / 256, 256, 0, stream>>>(s, d, n);
  };

  // ---- prep: weight conversion (fp32 -> bf16), x transpose, h = 0 ----
  conv(phi_x_W1, Wx1b, 512 * 64);
  conv(phi_x_W2, Wx2b, 512 * 512);
  conv(phi_z_W,  Wzb,  512 * 64);
  conv(enc_W1,   We1b, 512 * 1024);
  conv(enc_W2,   We2b, 512 * 512);
  conv(enc_mean_W, Wheb,            64 * 512);
  conv(enc_std_W,  Wheb + 64 * 512, 64 * 512);
  conv(prior_W,  Wpb,  512 * 512);
  conv(prior_mean_W, Whpb,            64 * 512);
  conv(prior_std_W,  Whpb + 64 * 512, 64 * 512);
  conv(gru_Wih,  Wihb, 1536 * 1024);
  conv(gru_Whh,  Whhb, 1536 * 512);
  xconv_kernel<<<(TB * D_ + 255) / 256, 256, 0, stream>>>(x, XB);
  zero_h_kernel<<<(B_ * H_) / 256, 256, 0, stream>>>(hF, hB);

  // ---- phase 1: phi_x for ALL timesteps (big parallel WMMA GEMMs) ----
  {
    dim3 g1(TB / 128, H_ / 64);
    gemm_bf16_kernel<1><<<g1, 128, 0, stream>>>(XB, XB, 64, 64, 64, 64,
        Wx1b, 64, phi_x_b1, nullptr, H1, 512);
    gemm_bf16_kernel<1><<<g1, 128, 0, stream>>>(H1, H1, 512, 512, 512, 512,
        Wx2b, 512, phi_x_b2, nullptr, PhiX, 512);
  }

  // ---- phase 2: recurrent chain, per-step WMMA pipeline ----
  const dim3 gH(B_ / 128, H_ / 64);    // (2, 8)
  const dim3 gG(B_ / 128, 1536 / 64);  // (2, 24)
  const dim3 gHd(B_ / 64, 2);          // (4, 2)
  for (int t = 0; t < T_; ++t) {
    const bf16* PhiXt = PhiX + (size_t)t * B_ * H_;
    float* emt = em + (size_t)t * B_ * Z_;
    float* est = es + (size_t)t * B_ * Z_;
    float* pmt = pm + (size_t)t * B_ * Z_;
    float* pst = ps + (size_t)t * B_ * Z_;

    // depend only on h (and precomputed PhiX) -> issue together at step start
    gemm_bf16_kernel<1><<<gH, 128, 0, stream>>>(PhiXt, hB, 512, 1024, 512, 512,
        We1b, 1024, enc_b1, nullptr, E1, 512);                       // enc layer 1
    gemm_bf16_kernel<1><<<gH, 128, 0, stream>>>(hB, hB, 512, 512, 512, 512,
        Wpb, 512, prior_b, nullptr, Pp, 512);                        // prior hidden
    gemm_bf16_kernel<0><<<gG, 128, 0, stream>>>(hB, hB, 512, 512, 512, 512,
        Whhb, 512, nullptr, Gh, nullptr, 1536);                      // gh = h @ Whh^T

    gemm_bf16_kernel<1><<<gH, 128, 0, stream>>>(E1, E1, 512, 512, 512, 512,
        We2b, 512, enc_b2, nullptr, E2, 512);                        // enc layer 2

    gemm_heads_kernel<<<gHd, 128, 0, stream>>>(Pp, Whpb,
        prior_mean_b, prior_std_b, pmt, pst);                        // prior heads
    gemm_heads_kernel<<<gHd, 128, 0, stream>>>(E2, Wheb,
        enc_mean_b, enc_std_b, emt, est);                            // enc heads

    zcalc_kernel<<<(B_ * Z_) / 256, 256, 0, stream>>>(
        emt, est, eps + (size_t)t * B_ * Z_, Zb);                    // z = eps*std+mean

    gemm_bf16_kernel<1><<<gH, 128, 0, stream>>>(Zb, Zb, 64, 64, 64, 64,
        Wzb, 64, phi_z_b, PZf, PZ, 512);                             // phi_z

    gemm_bf16_kernel<0><<<gG, 128, 0, stream>>>(PhiXt, PZ, 512, 1024, 512, 512,
        Wihb, 1024, nullptr, Gi, nullptr, 1536);                     // gi = [phi_x,phi_z]@Wih^T

    gru_cell_kernel<<<(B_ * H_) / 256, 256, 0, stream>>>(Gi, Gh, hF, hB);
  }

  final_kernel<<<(B_ * 2 * H_) / 256, 256, 0, stream>>>(PZf, hF, res);
}
